// GAT_31035433681287
// MI455X (gfx1250) — matrix operand order repaired and checked by reading, not compiled, not run on hardware
//
#include <hip/hip_runtime.h>
#include <hip/hip_bf16.h>
#include <math.h>

// ---------------------------------------------------------------------------
// Problem constants (from reference)
// ---------------------------------------------------------------------------
#define NNODES 20000
#define NEDGES 320000
#define NGRAPH 64
#define NHEAD  4
#define DIN    128
#define HID    64
#define MID    256
#define DOUTD  16
#define ETOT   (NEDGES + NNODES)   // self loops appended
#define GN_EPS 1e-5f
#define CHUNK_M 10                 // 1250 M-tiles = 125 blocks x 10
#define GEMM_WAVES 8               // 256-thread block = 8 wave32

typedef __attribute__((ext_vector_type(16))) __bf16 v16bf;
typedef __attribute__((ext_vector_type(8)))  float  v8f;

// ---------------------------------------------------------------------------
// Helpers
// ---------------------------------------------------------------------------
__device__ __forceinline__ unsigned short f32_to_bf16_rne(float f) {
    unsigned int u = __float_as_uint(f);
    unsigned int r = (u + 0x7FFFu + ((u >> 16) & 1u)) >> 16;
    return (unsigned short)r;
}

__device__ __forceinline__ void get_edge(const int* __restrict__ ei, int e,
                                         int& s, int& d) {
    if (e < NEDGES) { s = ei[e]; d = ei[NEDGES + e]; }
    else            { s = e - NEDGES; d = s; }
}

// float atomic max via signed/unsigned integer ordering trick
__device__ __forceinline__ void atomicMaxF(float* addr, float v) {
    if (v >= 0.0f) atomicMax((int*)addr, __float_as_int(v));
    else           atomicMin((unsigned int*)addr, __float_as_uint(v));
}

__device__ __forceinline__ float lrelu(float v, float slope) {
    return v > 0.0f ? v : slope * v;
}

// ---------------------------------------------------------------------------
// Utility kernels
// ---------------------------------------------------------------------------
__global__ void k_fill_f32(float* p, float v, long long n) {
    long long t = (long long)blockIdx.x * blockDim.x + threadIdx.x;
    if (t < n) p[t] = v;
}

__global__ void k_cvt_bf16(const float* __restrict__ in,
                           unsigned short* __restrict__ out, long long n) {
    long long t = (long long)blockIdx.x * blockDim.x + threadIdx.x;
    if (t < n) out[t] = f32_to_bf16_rne(in[t]);
}

// Pack fp32 weight W[K x FO] (row major) into WMMA B-operand layout:
//   packed[((kt*FO + n) * 32) + half*16 + j] = bf16(W[kt*32 + half*16 + j][n])
// so each lane reads its 16 K-values as one contiguous 32-byte chunk.
__global__ void k_pack_w(const float* __restrict__ W,
                         unsigned short* __restrict__ out, int K, int FO) {
    int t = blockIdx.x * blockDim.x + threadIdx.x;
    if (t >= K * FO) return;
    int k = t / FO, n = t % FO;
    int kt = k >> 5, r = k & 31, half = r >> 4, j = r & 15;
    out[((size_t)(kt * FO + n) << 5) + half * 16 + j] =
        f32_to_bf16_rne(W[(size_t)k * FO + n]);
}

// ---------------------------------------------------------------------------
// WMMA GEMM: C[N x FO] = A[N x K](bf16 row-major) @ Bpacked + bias
//
// One 256-thread block (8 wave32) stages the ENTIRE packed weight matrix in
// LDS (<=128KB bf16, WGP has 320KB), then sweeps CHUNK_M M-tiles x all
// N-tiles.  B-operand traffic is served from LDS (ds_load_b128), A from
// global with software prefetch.  EXEC stays all-ones inside each wave
// (all control flow is wave-uniform), as WMMA requires.
// ---------------------------------------------------------------------------
__global__ void k_gemm_wmma(const unsigned short* __restrict__ A,
                            const unsigned short* __restrict__ Bp,
                            const float* __restrict__ bias,
                            float* __restrict__ C,
                            int Nrows, int K, int FO) {
    extern __shared__ unsigned short smemW[];   // K*FO bf16, packed layout

    // ---- cooperative stage of packed weights into LDS (b128 granules) ----
    {
        const uint4* src = (const uint4*)Bp;
        uint4* dst = (uint4*)smemW;
        int nq = (K * FO) >> 3;                 // uint4 count (K*FO % 8 == 0)
        for (int i = threadIdx.x; i < nq; i += blockDim.x) dst[i] = src[i];
    }
    __syncthreads();

    const int waveId = threadIdx.x >> 5;
    const int lane   = threadIdx.x & 31;
    const int m      = lane & 15;
    const int half   = lane >> 4;
    const int tilesN = FO >> 4;
    const int tilesM = Nrows >> 4;
    const int jobs   = CHUNK_M * tilesN;
    const int ktn    = K >> 5;

    for (int job = waveId; job < jobs; job += GEMM_WAVES) {
        int tM = blockIdx.x * CHUNK_M + job / tilesN;
        int tN = job % tilesN;
        if (tM >= tilesM) continue;             // wave-uniform

        union { v16bf v; uint4 q[2]; } a, b;
        v8f acc = {0.f, 0.f, 0.f, 0.f, 0.f, 0.f, 0.f, 0.f};

        const unsigned short* Arow = A + (size_t)(tM * 16 + m) * K;

        // prefetch next job's A row (gfx1250 global_prefetch path)
        {
            int jn = job + GEMM_WAVES;
            if (jn < jobs) {
                int tMn = blockIdx.x * CHUNK_M + jn / tilesN;
                if (tMn < tilesM)
                    __builtin_prefetch(A + (size_t)(tMn * 16 + m) * K, 0, 1);
            }
        }

        for (int kt = 0; kt < ktn; ++kt) {
            int k0 = kt << 5;
            // A 16x32 bf16: lanes 0-15 hold K 0..15, 16-31 hold K 16..31
            a.q[0] = *(const uint4*)(Arow + k0 + 16 * half);
            a.q[1] = *(const uint4*)(Arow + k0 + 16 * half + 8);
            // B 32x16 bf16 from LDS: lane column = m, halves split K 0..15/16..31
            const unsigned short* bp =
                smemW + ((size_t)(kt * FO + tN * 16 + m) << 5) + half * 16;
            b.q[0] = *(const uint4*)(bp);
            b.q[1] = *(const uint4*)(bp + 8);
            acc = __builtin_amdgcn_wmma_f32_16x16x32_bf16(
                false, a.v, false, b.v, (short)0, acc, false, false);
        }
        // C layout: lane n = m, VGPR j holds row (j + 8*half)
        int n = tN * 16 + m;
        float bv = bias ? bias[n] : 0.0f;
        for (int j = 0; j < 8; ++j) {
            int row = tM * 16 + j + 8 * half;
            C[(size_t)row * FO + n] = acc[j] + bv;
        }
    }
}

// ---------------------------------------------------------------------------
// Edge-phase kernels (GATv2 attention)
// ---------------------------------------------------------------------------
__global__ void k_logits(const float* __restrict__ xl, const float* __restrict__ xr,
                         const float* __restrict__ att, const int* __restrict__ ei,
                         int FO, int C, float* __restrict__ logits,
                         float* __restrict__ mbuf) {
    long long t = (long long)blockIdx.x * blockDim.x + threadIdx.x;
    if (t >= (long long)ETOT * NHEAD) return;
    int e = (int)(t >> 2), h = (int)(t & 3);
    int s, d; get_edge(ei, e, s, d);
    const float* pl = xl + (size_t)s * FO + h * C;
    const float* pr = xr + (size_t)d * FO + h * C;
    const float* pa = att + h * C;
    float acc = 0.0f;
    for (int i = 0; i < C; ++i)
        acc += lrelu(pl[i] + pr[i], 0.2f) * pa[i];
    logits[(size_t)e * NHEAD + h] = acc;
    atomicMaxF(&mbuf[(size_t)d * NHEAD + h], acc);
}

__global__ void k_expsum(float* __restrict__ logits, const int* __restrict__ ei,
                         const float* __restrict__ mbuf, float* __restrict__ sbuf) {
    long long t = (long long)blockIdx.x * blockDim.x + threadIdx.x;
    if (t >= (long long)ETOT * NHEAD) return;
    int e = (int)(t >> 2), h = (int)(t & 3);
    int s, d; get_edge(ei, e, s, d); (void)s;
    float pe = __expf(logits[t] - mbuf[(size_t)d * NHEAD + h]);
    logits[t] = pe;                                   // overwrite with exp
    atomicAdd(&sbuf[(size_t)d * NHEAD + h], pe);
}

__global__ void k_aggregate(const float* __restrict__ pe, const float* __restrict__ sbuf,
                            const float* __restrict__ xl, const int* __restrict__ ei,
                            int FO, int C, float* __restrict__ g) {
    long long t = (long long)blockIdx.x * blockDim.x + threadIdx.x;
    if (t >= (long long)ETOT * FO) return;
    int e = (int)(t / FO), f = (int)(t % FO);
    int h = f / C;
    int s, d; get_edge(ei, e, s, d);
    float alpha = pe[(size_t)e * NHEAD + h] /
                  (sbuf[(size_t)d * NHEAD + h] + 1e-16f);
    atomicAdd(&g[(size_t)d * FO + f], alpha * xl[(size_t)s * FO + f]);
}

// h = leaky_relu(g + gat_bias + r, 0.01)
__global__ void k_combine(const float* __restrict__ g, const float* __restrict__ gb,
                          const float* __restrict__ r, float* __restrict__ h,
                          int FO) {
    long long t = (long long)blockIdx.x * blockDim.x + threadIdx.x;
    if (t >= (long long)NNODES * FO) return;
    int f = (int)(t % FO);
    h[t] = lrelu(g[t] + gb[f] + r[t], 0.01f);
}

// ---------------------------------------------------------------------------
// GraphNorm kernels
// ---------------------------------------------------------------------------
__global__ void k_gn_sum(const float* __restrict__ h, const int* __restrict__ batch,
                         int FO, float* __restrict__ gsum, float* __restrict__ gcnt) {
    long long t = (long long)blockIdx.x * blockDim.x + threadIdx.x;
    if (t >= (long long)NNODES * FO) return;
    int n = (int)(t / FO), f = (int)(t % FO);
    int b = batch[n];
    atomicAdd(&gsum[(size_t)b * FO + f], h[t]);
    if (f == 0) atomicAdd(&gcnt[b], 1.0f);
}

__global__ void k_gn_center(float* __restrict__ h, const int* __restrict__ batch,
                            const float* __restrict__ ms, int FO,
                            const float* __restrict__ gsum,
                            const float* __restrict__ gcnt,
                            float* __restrict__ gvar) {
    long long t = (long long)blockIdx.x * blockDim.x + threadIdx.x;
    if (t >= (long long)NNODES * FO) return;
    int n = (int)(t / FO), f = (int)(t % FO);
    int b = batch[n];
    float mean = gsum[(size_t)b * FO + f] / gcnt[b];
    float out = h[t] - ms[f] * mean;
    h[t] = out;
    atomicAdd(&gvar[(size_t)b * FO + f], out * out);
}

__global__ void k_gn_final(float* __restrict__ h, const int* __restrict__ batch,
                           const float* __restrict__ w, const float* __restrict__ bb,
                           int FO, const float* __restrict__ gvar,
                           const float* __restrict__ gcnt) {
    long long t = (long long)blockIdx.x * blockDim.x + threadIdx.x;
    if (t >= (long long)NNODES * FO) return;
    int n = (int)(t / FO), f = (int)(t % FO);
    int b = batch[n];
    float var = gvar[(size_t)b * FO + f] / gcnt[b];
    h[t] = w[f] * h[t] * rsqrtf(var + GN_EPS) + bb[f];
}

// ---------------------------------------------------------------------------
// Pooling + head
// ---------------------------------------------------------------------------
__global__ void k_pool_max(const float* __restrict__ h, const int* __restrict__ batch,
                           float* __restrict__ hg) {
    long long t = (long long)blockIdx.x * blockDim.x + threadIdx.x;
    if (t >= (long long)NNODES * HID) return;
    int n = (int)(t / HID), f = (int)(t % HID);
    atomicMaxF(&hg[(size_t)batch[n] * HID + f], h[t]);
}

__global__ void k_linear(const float* __restrict__ X, const float* __restrict__ W,
                         const float* __restrict__ b, float* __restrict__ O,
                         int R, int K, int M) {
    int t = blockIdx.x * blockDim.x + threadIdx.x;
    if (t >= R * M) return;
    int r = t / M, n = t % M;
    float acc = b[n];
    for (int k = 0; k < K; ++k) acc += X[(size_t)r * K + k] * W[(size_t)k * M + n];
    O[t] = acc;
}

__global__ void k_layernorm_lrelu(const float* __restrict__ o1,
                                  const float* __restrict__ w,
                                  const float* __restrict__ b,
                                  float* __restrict__ o2) {
    int r = blockIdx.x * blockDim.x + threadIdx.x;
    if (r >= NGRAPH) return;
    const float* row = o1 + (size_t)r * HID;
    float mu = 0.0f;
    for (int d = 0; d < HID; ++d) mu += row[d];
    mu /= HID;
    float var = 0.0f;
    for (int d = 0; d < HID; ++d) { float c = row[d] - mu; var += c * c; }
    var /= HID;
    float inv = rsqrtf(var + GN_EPS);
    for (int d = 0; d < HID; ++d) {
        float v = w[d] * (row[d] - mu) * inv + b[d];
        o2[(size_t)r * HID + d] = lrelu(v, 0.01f);
    }
}

// ---------------------------------------------------------------------------
// Host orchestration
// ---------------------------------------------------------------------------
static inline int gridFor(long long n, int block) {
    return (int)((n + block - 1) / block);
}

extern "C" void kernel_launch(void* const* d_in, const int* in_sizes, int n_in,
                              void* d_out, int out_size, void* d_ws, size_t ws_size,
                              hipStream_t stream) {
    (void)in_sizes; (void)n_in; (void)out_size; (void)ws_size;

    const float* x     = (const float*)d_in[0];
    const int*   ei    = (const int*)d_in[1];
    const int*   batch = (const int*)d_in[2];
    // params: 4 layers x 9 tensors, then fc1_W, fc1_b, ln_w, ln_b, fc2_W, fc2_b
    auto P = [&](int i) { return (const float*)d_in[3 + i]; };
    const float* fc1W = P(36); const float* fc1b = P(37);
    const float* lnw  = P(38); const float* lnb  = P(39);
    const float* fc2W = P(40); const float* fc2b = P(41);

    const int fiL[4] = {DIN, HID, MID, MID};
    const int foL[4] = {HID, MID, MID, HID};

    // ---- workspace carve-out (256B aligned) ----
    size_t off = 0;
    auto alloc = [&](size_t bytes) -> void* {
        off = (off + 255) & ~(size_t)255;
        void* p = (char*)d_ws + off;
        off += bytes;
        return p;
    };
    float*          hbuf = (float*)alloc((size_t)NNODES * MID * 4);
    unsigned short* habf = (unsigned short*)alloc((size_t)NNODES * MID * 2);
    float*          xl   = (float*)alloc((size_t)NNODES * MID * 4);
    float*          xr   = (float*)alloc((size_t)NNODES * MID * 4);
    float*          rr   = (float*)alloc((size_t)NNODES * MID * 4);
    float*          gg   = (float*)alloc((size_t)NNODES * MID * 4);
    float*          pe   = (float*)alloc((size_t)ETOT * NHEAD * 4);
    float*          mbuf = (float*)alloc((size_t)NNODES * NHEAD * 4);
    float*          sbuf = (float*)alloc((size_t)NNODES * NHEAD * 4);
    float*          gsum = (float*)alloc((size_t)NGRAPH * MID * 4);
    float*          gvar = (float*)alloc((size_t)NGRAPH * MID * 4);
    float*          gcnt = (float*)alloc((size_t)NGRAPH * 4);
    float*          hg   = (float*)alloc((size_t)NGRAPH * HID * 4);
    float*          o1   = (float*)alloc((size_t)NGRAPH * HID * 4);
    float*          o2   = (float*)alloc((size_t)NGRAPH * HID * 4);
    unsigned short* pWl  = (unsigned short*)alloc((size_t)MID * MID * 2);
    unsigned short* pWr  = (unsigned short*)alloc((size_t)MID * MID * 2);
    unsigned short* pWs  = (unsigned short*)alloc((size_t)MID * MID * 2);

    const int BLK = 256;

    // layer-1 input -> bf16
    {
        long long n = (long long)NNODES * DIN;
        k_cvt_bf16<<<gridFor(n, BLK), BLK, 0, stream>>>(x, habf, n);
    }

    for (int l = 0; l < 4; ++l) {
        const int fi = fiL[l], fo = foL[l], C = fo / NHEAD;
        const float* Wl   = P(l * 9 + 0);
        const float* Wr   = P(l * 9 + 1);
        const float* att  = P(l * 9 + 2);
        const float* gb   = P(l * 9 + 3);
        const float* Wres = P(l * 9 + 4);
        const float* bres = P(l * 9 + 5);
        const float* nw   = P(l * 9 + 6);
        const float* nb   = P(l * 9 + 7);
        const float* nms  = P(l * 9 + 8);

        // pack weights into WMMA B layout
        int we = fi * fo;
        k_pack_w<<<gridFor(we, BLK), BLK, 0, stream>>>(Wl, pWl, fi, fo);
        k_pack_w<<<gridFor(we, BLK), BLK, 0, stream>>>(Wr, pWr, fi, fo);
        k_pack_w<<<gridFor(we, BLK), BLK, 0, stream>>>(Wres, pWs, fi, fo);

        // dense transforms via bf16 WMMA (f32 accumulate), weights in LDS
        int tilesM = NNODES / 16;                       // 1250
        int gemmGrid = (tilesM + CHUNK_M - 1) / CHUNK_M; // 125 blocks
        size_t ldsB = (size_t)fi * fo * 2;              // packed weights
        k_gemm_wmma<<<gemmGrid, BLK, ldsB, stream>>>(habf, pWl, nullptr, xl, NNODES, fi, fo);
        k_gemm_wmma<<<gemmGrid, BLK, ldsB, stream>>>(habf, pWr, nullptr, xr, NNODES, fi, fo);
        k_gemm_wmma<<<gemmGrid, BLK, ldsB, stream>>>(habf, pWs, bres,    rr, NNODES, fi, fo);

        // init segment buffers
        long long nh = (long long)NNODES * NHEAD;
        k_fill_f32<<<gridFor(nh, BLK), BLK, 0, stream>>>(mbuf, -INFINITY, nh);
        k_fill_f32<<<gridFor(nh, BLK), BLK, 0, stream>>>(sbuf, 0.0f, nh);
        long long nf = (long long)NNODES * fo;
        k_fill_f32<<<gridFor(nf, BLK), BLK, 0, stream>>>(gg, 0.0f, nf);

        // attention
        long long eh = (long long)ETOT * NHEAD;
        k_logits<<<gridFor(eh, BLK), BLK, 0, stream>>>(xl, xr, att, ei, fo, C, pe, mbuf);
        k_expsum<<<gridFor(eh, BLK), BLK, 0, stream>>>(pe, ei, mbuf, sbuf);
        long long ef = (long long)ETOT * fo;
        k_aggregate<<<gridFor(ef, BLK), BLK, 0, stream>>>(pe, sbuf, xl, ei, fo, C, gg);

        // residual + activation
        k_combine<<<gridFor(nf, BLK), BLK, 0, stream>>>(gg, gb, rr, hbuf, fo);

        // GraphNorm
        long long bf = (long long)NGRAPH * fo;
        k_fill_f32<<<gridFor(bf, BLK), BLK, 0, stream>>>(gsum, 0.0f, bf);
        k_fill_f32<<<gridFor(bf, BLK), BLK, 0, stream>>>(gvar, 0.0f, bf);
        k_fill_f32<<<gridFor(NGRAPH, BLK), BLK, 0, stream>>>(gcnt, 0.0f, NGRAPH);
        k_gn_sum<<<gridFor(nf, BLK), BLK, 0, stream>>>(hbuf, batch, fo, gsum, gcnt);
        k_gn_center<<<gridFor(nf, BLK), BLK, 0, stream>>>(hbuf, batch, nms, fo, gsum, gcnt, gvar);
        k_gn_final<<<gridFor(nf, BLK), BLK, 0, stream>>>(hbuf, batch, nw, nb, fo, gvar, gcnt);

        // next-layer bf16 activations
        k_cvt_bf16<<<gridFor(nf, BLK), BLK, 0, stream>>>(hbuf, habf, nf);
    }

    // global max pool
    long long bh = (long long)NGRAPH * HID;
    k_fill_f32<<<gridFor(bh, BLK), BLK, 0, stream>>>(hg, -INFINITY, bh);
    long long nh64 = (long long)NNODES * HID;
    k_pool_max<<<gridFor(nh64, BLK), BLK, 0, stream>>>(hbuf, batch, hg);

    // head: fc1 -> LN -> leaky -> fc2
    k_linear<<<gridFor(NGRAPH * HID, BLK), BLK, 0, stream>>>(hg, fc1W, fc1b, o1, NGRAPH, HID, HID);
    k_layernorm_lrelu<<<gridFor(NGRAPH, BLK), BLK, 0, stream>>>(o1, lnw, lnb, o2);
    k_linear<<<gridFor(NGRAPH * DOUTD, BLK), BLK, 0, stream>>>(o2, fc2W, fc2b, (float*)d_out,
                                                               NGRAPH, HID, DOUTD);
}
